// CompositionNet_6210522710675
// MI455X (gfx1250) — compile-verified, matrix operand order checked by reading, not run on hardware
//
#include <hip/hip_runtime.h>
#include <hip/hip_bf16.h>
#include <math.h>

// ---------------------------------------------------------------------------
// CompositionNet (CGCNN-style) forward for MI455X / gfx1250, wave32 + WMMA.
// Edge GEMM [1.6M x 152] @ [152 x 128] done twice per layer (BN stats pass +
// apply pass) with v_wmma_f32_16x16x32_f16; atom_fea gathers hit L2 (12.8MB),
// bond_fea streams from HBM (prefetched).  Branch-free fragment loaders;
// B-fragments preloaded per K-step so wmma bursts overlap the load clause.
// ---------------------------------------------------------------------------

typedef __attribute__((ext_vector_type(16))) _Float16 v16h;
typedef __attribute__((ext_vector_type(8)))  float    v8f;

#define N_ATOMS 100000
#define M_EDGES 1600000
#define C_CRYS  10000
#define ORIG_D  200
#define NBR_D   24
#define ATOM_D  64
#define H_D     128
#define KE_PAD  160   // 2*64+24=152 padded to multiple of 32
#define KO_PAD  224   // 200 padded to multiple of 32
#define BN_EPS  1e-5f

// ---------------- device helpers ----------------

__device__ __forceinline__ float softplusf(float x) {
  return (x > 20.f) ? x : log1pf(expf(x));
}
__device__ __forceinline__ float sigmoidf_(float x) {
  return 1.f / (1.f + expf(-x));
}

// K offset (0..31) held by element j (0..15) of a v16h A/B fragment in `lane`.
// Per CDNA5 ISA 16-bit A 16x32 layout: VGPR v holds K = (v/4)*16 + (lane/16)*8
// + (v%4)*2 + e, for packed halves e=0,1.  (B assumed mirrored, n = lane%16.)
__device__ __forceinline__ int frag_k(int j, int lane) {
  int v = j >> 1, e = j & 1;
  return ((v >> 2) << 4) + ((lane >> 4) << 3) + ((v & 3) << 1) + e;
}

// Unconditional contiguous loader: per lane two 16B chunks -> global_load_b128.
__device__ __forceinline__ v16h load_frag_contig(const _Float16* __restrict__ src,
                                                 int kbase, int lane) {
  v16h a;
#pragma unroll
  for (int j = 0; j < 16; ++j)
    a[j] = src[kbase + frag_k(j, lane)];
  return a;
}

// Bond K-step (K=128..159 of the edge row): valid for k<24, rest is zero pad.
// Load is unconditional (overreads stay inside the workspace), zero via select.
__device__ __forceinline__ v16h load_frag_bond(const _Float16* __restrict__ bondrow,
                                               int lane) {
  v16h a;
#pragma unroll
  for (int j = 0; j < 16; ++j) {
    int k = frag_k(j, lane);
    _Float16 t = bondrow[k];
    a[j] = (k < NBR_D) ? t : (_Float16)0.f;
  }
  return a;
}

// B fragment from transposed/padded fp16 weights Wt[n][k], row stride kpad.
__device__ __forceinline__ v16h load_Bfrag(const _Float16* __restrict__ Wt,
                                           int ncol0, int kbase, int kpad, int lane) {
  int n = ncol0 + (lane & 15);
  v16h b;
#pragma unroll
  for (int j = 0; j < 16; ++j)
    b[j] = Wt[n * kpad + kbase + frag_k(j, lane)];
  return b;
}

// ---------------- prep / utility kernels ----------------

__global__ void zero_f32(float* p, int n) {
  int i = blockIdx.x * 256 + threadIdx.x;
  if (i < n) p[i] = 0.f;
}

__global__ void convert_bond(const float* __restrict__ nbr, _Float16* __restrict__ bond16, int n) {
  int i = blockIdx.x * 256 + threadIdx.x;
  if (i < n) bond16[i] = (_Float16)nbr[i];
}

__global__ void count_edges(const int* __restrict__ selfIdx, float* cnt, int m) {
  int i = blockIdx.x * 256 + threadIdx.x;
  if (i < m) atomicAdd(&cnt[selfIdx[i]], 1.f);
}

// Repack edge weights: Wt[n][k] fp16 (n: 0..63 filter, 64..127 core), pad K.
// Also pack per-layer params: eparam[0..127]=bias, [128..255]=gamma, [256..383]=beta.
__global__ void prep_edge_w(const float* __restrict__ fW, const float* __restrict__ cW,
                            const float* __restrict__ fb, const float* __restrict__ cb,
                            const float* __restrict__ gF, const float* __restrict__ bF,
                            const float* __restrict__ gC, const float* __restrict__ bC,
                            _Float16* __restrict__ Wt, float* __restrict__ eparam) {
  int i = blockIdx.x * 256 + threadIdx.x;
  if (i < H_D * KE_PAD) {
    int n = i / KE_PAD, k = i % KE_PAD;
    float v = 0.f;
    if (k < 152) v = (n < 64) ? fW[k * 64 + n] : cW[k * 64 + (n - 64)];
    Wt[i] = (_Float16)v;
  }
  if (i < H_D) {
    eparam[i]       = (i < 64) ? fb[i] : cb[i - 64];
    eparam[128 + i] = (i < 64) ? gF[i] : gC[i - 64];
    eparam[256 + i] = (i < 64) ? bF[i] : bC[i - 64];
  }
}

__global__ void prep_emb_w(const float* __restrict__ embW, _Float16* __restrict__ Wt) {
  int i = blockIdx.x * 256 + threadIdx.x;  // 64 * 224
  if (i < ATOM_D * KO_PAD) {
    int n = i / KO_PAD, k = i % KO_PAD;
    Wt[i] = (_Float16)((k < ORIG_D) ? embW[k * ATOM_D + n] : 0.f);
  }
}

__global__ void prep_fc_w(const float* __restrict__ gfcW, _Float16* __restrict__ Wt) {
  int i = blockIdx.x * 256 + threadIdx.x;  // 128 * 64
  if (i < H_D * ATOM_D) {
    int n = i / ATOM_D, k = i % ATOM_D;
    Wt[i] = (_Float16)gfcW[k * H_D + n];
  }
}

// BN finalize: scale = g*rsqrt(var+eps), shift = beta - scale*mu (+ scale*bias).
__global__ void finalize_stats(const float* __restrict__ gsum, const float* __restrict__ gsq,
                               const float* __restrict__ gamma, const float* __restrict__ beta,
                               const float* __restrict__ bias,  // may be null
                               float* __restrict__ scale, float* __restrict__ shift,
                               float invB, int d) {
  int j = threadIdx.x;
  if (j >= d) return;
  float mu  = gsum[j] * invB;
  float var = gsq[j] * invB - mu * mu;
  float sc  = gamma[j] * rsqrtf(var + BN_EPS);
  float sh  = beta[j] - sc * mu;
  if (bias) sh += sc * bias[j];
  scale[j] = sc;
  shift[j] = sh;
}

// ---------------- embedding GEMM: [N,200]@[200,64]+b ----------------

__global__ __launch_bounds__(256)
void embed_gemm(const float* __restrict__ orig, const _Float16* __restrict__ Wt,
                const float* __restrict__ bias, float* __restrict__ atomf,
                _Float16* __restrict__ atom16, int n_rows) {
  int tid = threadIdx.x, lane = tid & 31, wave = tid >> 5;
  int rowbase = blockIdx.x * 128 + wave * 16;
  int arow = rowbase + (lane & 15);
  int arowc = arow < n_rows ? arow : n_rows - 1;      // clamp loads, mask stores
  const float* __restrict__ orow = orig + arowc * ORIG_D;
  v8f acc[4] = {};
#pragma unroll
  for (int ks = 0; ks < KO_PAD / 32; ++ks) {
    v16h a;
    if (ks < 6) {            // k = ks*32 + frag_k <= 191 < 200: always valid
#pragma unroll
      for (int j = 0; j < 16; ++j)
        a[j] = (_Float16)orow[ks * 32 + frag_k(j, lane)];
    } else {                 // k = 192 + frag_k: valid iff frag_k < 8
#pragma unroll
      for (int j = 0; j < 16; ++j) {
        int k = frag_k(j, lane);
        float t = orow[192 + (k & 7)];                // always in-bounds
        a[j] = (k < 8) ? (_Float16)t : (_Float16)0.f;
      }
    }
    v16h b[4];
#pragma unroll
    for (int nt = 0; nt < 4; ++nt)
      b[nt] = load_Bfrag(Wt, nt * 16, ks * 32, KO_PAD, lane);
#pragma unroll
    for (int nt = 0; nt < 4; ++nt)
      acc[nt] = __builtin_amdgcn_wmma_f32_16x16x32_f16(false, a, false, b[nt], (short)0,
                                                       acc[nt], false, false);
  }
  int rbase = rowbase + ((lane >> 4) << 3);
#pragma unroll
  for (int nt = 0; nt < 4; ++nt) {
    int col = (lane & 15) + nt * 16;
    float bv = bias[col];
#pragma unroll
    for (int r = 0; r < 8; ++r) {
      int row = rbase + r;
      if (row < n_rows) {
        float y = acc[nt][r] + bv;
        atomf[row * ATOM_D + col]  = y;
        atom16[row * ATOM_D + col] = (_Float16)y;
      }
    }
  }
}

// ---------------- shared edge-GEMM matmul body ----------------

__device__ __forceinline__ void edge_matmul(const _Float16* __restrict__ atom16,
                                            const _Float16* __restrict__ bond16,
                                            int s, int nb, int erow,
                                            const _Float16* __restrict__ Wt,
                                            int lane, v8f acc[8]) {
  const _Float16* __restrict__ srow = atom16 + s * ATOM_D;
  const _Float16* __restrict__ nrow = atom16 + nb * ATOM_D;
  const _Float16* __restrict__ brow = bond16 + erow * NBR_D;
#pragma unroll
  for (int ks = 0; ks < KE_PAD / 32; ++ks) {
    v16h a;
    if (ks == 0)      a = load_frag_contig(srow, 0, lane);
    else if (ks == 1) a = load_frag_contig(srow, 32, lane);
    else if (ks == 2) a = load_frag_contig(nrow, 0, lane);
    else if (ks == 3) a = load_frag_contig(nrow, 32, lane);
    else              a = load_frag_bond(brow, lane);
    // Preload all 8 B-fragments: one big load clause, then a wmma burst with
    // partial s_wait_loadcnt overlap instead of load->wait0->wmma serialization.
    v16h b[8];
#pragma unroll
    for (int nt = 0; nt < 8; ++nt)
      b[nt] = load_Bfrag(Wt, nt * 16, ks * 32, KE_PAD, lane);
#pragma unroll
    for (int nt = 0; nt < 8; ++nt)
      acc[nt] = __builtin_amdgcn_wmma_f32_16x16x32_f16(false, a, false, b[nt], (short)0,
                                                       acc[nt], false, false);
  }
}

// Prefetch next block's streaming bond rows (the only HBM-bound operand).
__device__ __forceinline__ void prefetch_bond(const _Float16* __restrict__ bond16,
                                              int tid, int m) {
  int idx = (blockIdx.x + 1) * 128 * NBR_D + tid * 12;   // 128*24 halfs / 256 thr
  if (idx < m * NBR_D) __builtin_prefetch(bond16 + idx, 0, 1);
}

// ---------------- edge GEMM pass 1: BN column stats ----------------

__global__ __launch_bounds__(256)
void edge_gemm_stats(const _Float16* __restrict__ atom16, const _Float16* __restrict__ bond16,
                     const int* __restrict__ selfIdx, const int* __restrict__ nbrIdx,
                     const _Float16* __restrict__ Wt, const float* __restrict__ eparam,
                     float* __restrict__ gsum, float* __restrict__ gsq, int m) {
  __shared__ float lsum[H_D], lsq[H_D];
  int tid = threadIdx.x;
  if (tid < H_D) { lsum[tid] = 0.f; lsq[tid] = 0.f; }
  __syncthreads();
  prefetch_bond(bond16, tid, m);
  int lane = tid & 31, wave = tid >> 5;
  int rowbase = blockIdx.x * 128 + wave * 16;
  int arow = rowbase + (lane & 15);
  int arowc = arow < m ? arow : m - 1;
  int s = selfIdx[arowc], nb = nbrIdx[arowc];
  v8f acc[8] = {};
  edge_matmul(atom16, bond16, s, nb, arowc, Wt, lane, acc);
  int rbase = rowbase + ((lane >> 4) << 3);
#pragma unroll
  for (int nt = 0; nt < 8; ++nt) {
    int col = (lane & 15) + nt * 16;
    float bv = eparam[col];
    float sm = 0.f, sq = 0.f;
#pragma unroll
    for (int r = 0; r < 8; ++r) {
      if (rbase + r < m) {
        float y = acc[nt][r] + bv;
        sm += y; sq += y * y;
      }
    }
    atomicAdd(&lsum[col], sm);
    atomicAdd(&lsq[col], sq);
  }
  __syncthreads();
  if (tid < H_D) {
    atomicAdd(&gsum[tid], lsum[tid]);
    atomicAdd(&gsq[tid],  lsq[tid]);
  }
}

// ---------------- edge GEMM pass 2: BN apply + gate + scatter-add ----------------

__global__ __launch_bounds__(256)
void edge_gemm_apply(const _Float16* __restrict__ atom16, const _Float16* __restrict__ bond16,
                     const int* __restrict__ selfIdx, const int* __restrict__ nbrIdx,
                     const _Float16* __restrict__ Wt, const float* __restrict__ scale,
                     const float* __restrict__ shift, float* __restrict__ summed, int m) {
  int tid = threadIdx.x, lane = tid & 31, wave = tid >> 5;
  prefetch_bond(bond16, tid, m);
  int rowbase = blockIdx.x * 128 + wave * 16;
  int arow = rowbase + (lane & 15);
  int arowc = arow < m ? arow : m - 1;
  int s = selfIdx[arowc], nb = nbrIdx[arowc];
  v8f acc[8] = {};
  edge_matmul(atom16, bond16, s, nb, arowc, Wt, lane, acc);
  int rbase = rowbase + ((lane >> 4) << 3);
  int dst[8];
#pragma unroll
  for (int r = 0; r < 8; ++r) {
    int row = rbase + r;
    dst[r] = selfIdx[row < m ? row : m - 1];
  }
  // filter col j lives in tile nt, core col j+64 in tile nt+4: same lane, same VGPR.
#pragma unroll
  for (int nt = 0; nt < 4; ++nt) {
    int j = (lane & 15) + nt * 16;
    float scF = scale[j],      shF = shift[j];
    float scC = scale[j + 64], shC = shift[j + 64];
#pragma unroll
    for (int r = 0; r < 8; ++r) {
      if (rbase + r < m) {
        float f = sigmoidf_(scF * acc[nt][r] + shF);
        float c = softplusf(scC * acc[nt + 4][r] + shC);
        atomicAdd(&summed[dst[r] * ATOM_D + j], f * c);
      }
    }
  }
}

// ---------------- node-side kernels ----------------

__global__ __launch_bounds__(256)
void node_stats(float* __restrict__ summed, const float* __restrict__ cnt,
                float* __restrict__ gsum, float* __restrict__ gsq, int total) {
  __shared__ float lsum[ATOM_D], lsq[ATOM_D];
  int tid = threadIdx.x;
  if (tid < ATOM_D) { lsum[tid] = 0.f; lsq[tid] = 0.f; }
  __syncthreads();
  int i = blockIdx.x * 256 + tid;
  if (i < total) {
    int n = i >> 6, j = i & 63;
    float v = summed[i] / fmaxf(cnt[n], 1.f);
    summed[i] = v;
    atomicAdd(&lsum[j], v);
    atomicAdd(&lsq[j], v * v);
  }
  __syncthreads();
  if (tid < ATOM_D) {
    atomicAdd(&gsum[tid], lsum[tid]);
    atomicAdd(&gsq[tid],  lsq[tid]);
  }
}

__global__ void node_update(float* __restrict__ atomf, _Float16* __restrict__ atom16,
                            const float* __restrict__ summed,
                            const float* __restrict__ scale, const float* __restrict__ shift,
                            int total) {
  int i = blockIdx.x * 256 + threadIdx.x;
  if (i < total) {
    int j = i & 63;
    float v = softplusf(atomf[i] + scale[j] * summed[i] + shift[j]);
    atomf[i]  = v;
    atom16[i] = (_Float16)v;
  }
}

// ---------------- crystal pooling + head ----------------

__global__ void crys_accum(const float* __restrict__ atomf, const int* __restrict__ crysIdx,
                           float* __restrict__ csum, float* __restrict__ ccnt, int total) {
  int i = blockIdx.x * 256 + threadIdx.x;
  if (i < total) {
    int n = i >> 6, j = i & 63;
    int c = crysIdx[n];
    atomicAdd(&csum[c * ATOM_D + j], atomf[i]);
    if (j == 0) atomicAdd(&ccnt[c], 1.f);
  }
}

__global__ void crys_final(const float* __restrict__ csum, const float* __restrict__ ccnt,
                           _Float16* __restrict__ crys16, int total) {
  int i = blockIdx.x * 256 + threadIdx.x;
  if (i < total) {
    float g = softplusf(csum[i] / fmaxf(ccnt[i >> 6], 1.f));
    crys16[i] = (_Float16)g;
  }
}

// [C,64]@[64,128]+b with column stats (z stored fp32 for the BN apply pass).
__global__ __launch_bounds__(256)
void fc_gemm(const _Float16* __restrict__ crys16, const _Float16* __restrict__ Wt,
             const float* __restrict__ bias, float* __restrict__ zbuf,
             float* __restrict__ gsum, float* __restrict__ gsq, int c_rows) {
  __shared__ float lsum[H_D], lsq[H_D];
  int tid = threadIdx.x;
  if (tid < H_D) { lsum[tid] = 0.f; lsq[tid] = 0.f; }
  __syncthreads();
  int lane = tid & 31, wave = tid >> 5;
  int rowbase = blockIdx.x * 128 + wave * 16;
  int arow = rowbase + (lane & 15);
  int arowc = arow < c_rows ? arow : c_rows - 1;   // clamp loads, mask stores/stats
  const _Float16* __restrict__ crow = crys16 + arowc * ATOM_D;
  v8f acc[8] = {};
#pragma unroll
  for (int ks = 0; ks < 2; ++ks) {
    v16h a = load_frag_contig(crow, ks * 32, lane);
    v16h b[8];
#pragma unroll
    for (int nt = 0; nt < 8; ++nt)
      b[nt] = load_Bfrag(Wt, nt * 16, ks * 32, ATOM_D, lane);
#pragma unroll
    for (int nt = 0; nt < 8; ++nt)
      acc[nt] = __builtin_amdgcn_wmma_f32_16x16x32_f16(false, a, false, b[nt], (short)0,
                                                       acc[nt], false, false);
  }
  int rbase = rowbase + ((lane >> 4) << 3);
#pragma unroll
  for (int nt = 0; nt < 8; ++nt) {
    int col = (lane & 15) + nt * 16;
    float bv = bias[col];
    float sm = 0.f, sq = 0.f;
#pragma unroll
    for (int r = 0; r < 8; ++r) {
      int row = rbase + r;
      if (row < c_rows) {
        float z = acc[nt][r] + bv;
        zbuf[row * H_D + col] = z;
        sm += z; sq += z * z;
      }
    }
    atomicAdd(&lsum[col], sm);
    atomicAdd(&lsq[col], sq);
  }
  __syncthreads();
  if (tid < H_D) {
    atomicAdd(&gsum[tid], lsum[tid]);
    atomicAdd(&gsq[tid],  lsq[tid]);
  }
}

__global__ void fc_out_kernel(const float* __restrict__ zbuf,
                              const float* __restrict__ scale, const float* __restrict__ shift,
                              const float* __restrict__ fcW, const float* __restrict__ fcb,
                              float* __restrict__ out, int c_rows) {
  int c = blockIdx.x * 256 + threadIdx.x;
  if (c >= c_rows) return;
  float acc = fcb[0];
#pragma unroll 4
  for (int j = 0; j < H_D; ++j)
    acc += softplusf(scale[j] * zbuf[c * H_D + j] + shift[j]) * fcW[j];
  out[c] = acc;
}

// ---------------- host launcher ----------------

extern "C" void kernel_launch(void* const* d_in, const int* in_sizes, int n_in,
                              void* d_out, int out_size, void* d_ws, size_t ws_size,
                              hipStream_t stream) {
  (void)in_sizes; (void)n_in; (void)out_size; (void)ws_size;
  // ---- inputs (jax pytree flatten order: top-level dict insertion order,
  //      params dict keys sorted alphabetically, depth-first) ----
  const float* orig    = (const float*)d_in[0];   // [N,200]
  const float* nbr     = (const float*)d_in[1];   // [M,24]
  const int*   selfIdx = (const int*)d_in[2];     // [M]
  const int*   nbrIdx  = (const int*)d_in[3];     // [M]
  const int*   crysIdx = (const int*)d_in[4];     // [N]
  const float* embW    = (const float*)d_in[5];   // embedding.W [200,64]
  const float* embB    = (const float*)d_in[6];   // embedding.b [64]
  const float* fcoW    = (const float*)d_in[7];   // fc_out.W [128,1]
  const float* fcoB    = (const float*)d_in[8];   // fc_out.b [1]
  const float* gfcW    = (const float*)d_in[9];   // graph_to_fc.W [64,128]
  const float* gfcB    = (const float*)d_in[10];  // graph_to_fc.b [128]
  const float* gfcBNb  = (const float*)d_in[11];  // graph_to_fc_bn.b
  const float* gfcBNg  = (const float*)d_in[12];  // graph_to_fc_bn.g
  // graphs[l] leaves at base 13+10*l:
  // +0 bn_core.b +1 bn_core.g +2 bn_filter.b +3 bn_filter.g
  // +4 bn_output.b +5 bn_output.g +6 core.W +7 core.b +8 filter.W +9 filter.b

  // ---- workspace carve-up ----
  char* w = (char*)d_ws;
  size_t off = 0;
  auto alloc = [&](size_t bytes) -> void* {
    void* p = w + off;
    off = (off + bytes + 255) & ~(size_t)255;
    return p;
  };
  float*    atomf   = (float*)alloc((size_t)N_ATOMS * ATOM_D * 4);
  _Float16* atom16  = (_Float16*)alloc((size_t)N_ATOMS * ATOM_D * 2);
  _Float16* bond16  = (_Float16*)alloc((size_t)M_EDGES * NBR_D * 2);
  float*    summed  = (float*)alloc((size_t)N_ATOMS * ATOM_D * 4);
  float*    cnt     = (float*)alloc((size_t)N_ATOMS * 4);
  float*    estats  = (float*)alloc(512 * 4);  // sum|sq|scale|shift (128 each)
  float*    nstats  = (float*)alloc(256 * 4);  // sum|sq|scale|shift (64 each)
  float*    fstats  = (float*)alloc(512 * 4);
  _Float16* wedge0  = (_Float16*)alloc((size_t)H_D * KE_PAD * 2);
  _Float16* wedge1  = (_Float16*)alloc((size_t)H_D * KE_PAD * 2);
  _Float16* wedge2  = (_Float16*)alloc((size_t)H_D * KE_PAD * 2);
  float*    eparam0 = (float*)alloc(384 * 4);  // bias|gamma|beta (128 each)
  float*    eparam1 = (float*)alloc(384 * 4);
  float*    eparam2 = (float*)alloc(384 * 4);
  _Float16* wemb    = (_Float16*)alloc((size_t)ATOM_D * KO_PAD * 2);
  _Float16* wfc     = (_Float16*)alloc((size_t)H_D * ATOM_D * 2);
  float*    csum    = (float*)alloc((size_t)C_CRYS * ATOM_D * 4);
  float*    ccnt    = (float*)alloc((size_t)C_CRYS * 4);
  _Float16* crys16  = (_Float16*)alloc((size_t)C_CRYS * ATOM_D * 2);
  float*    zbuf    = (float*)alloc((size_t)C_CRYS * H_D * 4);

  _Float16* wedge[3]  = {wedge0, wedge1, wedge2};
  float*    eparam[3] = {eparam0, eparam1, eparam2};

  dim3 blk(256);
  auto gr = [](int n) { return dim3((unsigned)((n + 255) / 256)); };

  // ---- weight repack ----
  for (int l = 0; l < 3; ++l) {
    int base = 13 + 10 * l;
    prep_edge_w<<<gr(H_D * KE_PAD), blk, 0, stream>>>(
        (const float*)d_in[base + 8], (const float*)d_in[base + 6],   // filter.W, core.W
        (const float*)d_in[base + 9], (const float*)d_in[base + 7],   // filter.b, core.b
        (const float*)d_in[base + 3], (const float*)d_in[base + 2],   // bn_filter.g, .b
        (const float*)d_in[base + 1], (const float*)d_in[base + 0],   // bn_core.g, .b
        wedge[l], eparam[l]);
  }
  prep_emb_w<<<gr(ATOM_D * KO_PAD), blk, 0, stream>>>(embW, wemb);
  prep_fc_w<<<gr(H_D * ATOM_D), blk, 0, stream>>>(gfcW, wfc);
  convert_bond<<<gr(M_EDGES * NBR_D), blk, 0, stream>>>(nbr, bond16, M_EDGES * NBR_D);

  // ---- edge counts (constant across layers) ----
  zero_f32<<<gr(N_ATOMS), blk, 0, stream>>>(cnt, N_ATOMS);
  count_edges<<<gr(M_EDGES), blk, 0, stream>>>(selfIdx, cnt, M_EDGES);

  // ---- embedding ----
  embed_gemm<<<dim3((N_ATOMS + 127) / 128), blk, 0, stream>>>(orig, wemb, embB,
                                                              atomf, atom16, N_ATOMS);

  // ---- 3 message-passing layers ----
  const int NA = N_ATOMS * ATOM_D;
  for (int l = 0; l < 3; ++l) {
    int base = 13 + 10 * l;
    zero_f32<<<gr(512), blk, 0, stream>>>(estats, 512);
    zero_f32<<<gr(256), blk, 0, stream>>>(nstats, 256);
    zero_f32<<<gr(NA), blk, 0, stream>>>(summed, NA);

    edge_gemm_stats<<<dim3((M_EDGES + 127) / 128), blk, 0, stream>>>(
        atom16, bond16, selfIdx, nbrIdx, wedge[l], eparam[l],
        estats + 0, estats + 128, M_EDGES);
    finalize_stats<<<1, 128, 0, stream>>>(estats + 0, estats + 128,
        eparam[l] + 128, eparam[l] + 256, eparam[l] + 0,
        estats + 256, estats + 384, 1.f / (float)M_EDGES, 128);
    edge_gemm_apply<<<dim3((M_EDGES + 127) / 128), blk, 0, stream>>>(
        atom16, bond16, selfIdx, nbrIdx, wedge[l],
        estats + 256, estats + 384, summed, M_EDGES);

    node_stats<<<gr(NA), blk, 0, stream>>>(summed, cnt, nstats + 0, nstats + 64, NA);
    finalize_stats<<<1, 64, 0, stream>>>(nstats + 0, nstats + 64,
        (const float*)d_in[base + 5], (const float*)d_in[base + 4],  // bn_output.g, .b
        nullptr, nstats + 128, nstats + 192, 1.f / (float)N_ATOMS, 64);
    node_update<<<gr(NA), blk, 0, stream>>>(atomf, atom16, summed,
                                            nstats + 128, nstats + 192, NA);
  }

  // ---- crystal pooling + head ----
  const int CA = C_CRYS * ATOM_D;
  zero_f32<<<gr(CA), blk, 0, stream>>>(csum, CA);
  zero_f32<<<gr(C_CRYS), blk, 0, stream>>>(ccnt, C_CRYS);
  zero_f32<<<gr(512), blk, 0, stream>>>(fstats, 512);
  crys_accum<<<gr(NA), blk, 0, stream>>>(atomf, crysIdx, csum, ccnt, NA);
  crys_final<<<gr(CA), blk, 0, stream>>>(csum, ccnt, crys16, CA);

  fc_gemm<<<dim3((C_CRYS + 127) / 128), blk, 0, stream>>>(
      crys16, wfc, gfcB, zbuf, fstats + 0, fstats + 128, C_CRYS);
  finalize_stats<<<1, 128, 0, stream>>>(fstats + 0, fstats + 128,
      gfcBNg, gfcBNb, nullptr, fstats + 256, fstats + 384, 1.f / (float)C_CRYS, 128);
  fc_out_kernel<<<gr(C_CRYS), blk, 0, stream>>>(zbuf, fstats + 256, fstats + 384,
                                                fcoW, fcoB, (float*)d_out, C_CRYS);
}
